// WeightShareSelfAttention_65704409694877
// MI455X (gfx1250) — matrix operand order repaired and checked by reading, not compiled, not run on hardware
//
#include <hip/hip_runtime.h>
#include <hip/hip_bf16.h>
#include <stdint.h>

typedef __attribute__((ext_vector_type(16))) __bf16 v16bf;
typedef __attribute__((ext_vector_type(8)))  float  v8f;
typedef unsigned short u16;
typedef unsigned int   u32;
typedef unsigned long long u64;

#define D_MODEL 1024
#define L_TOT   2048
#define NH      16
#define DH      64
#define QKV_M   3072
#define ATTN_SCALE 0.125f
#define LN_EPS  1e-5f

#define HAVE_ASYNC_LDS __has_builtin(__builtin_amdgcn_global_load_async_to_lds_b128)
#define HAVE_TDM       __has_builtin(__builtin_amdgcn_tensor_load_to_lds)

union Frag {
  uint4 q[2];
  v16bf v;
};

__device__ __forceinline__ u16 f2bf(float f) {
  u32 u = __float_as_uint(f);
  u += 0x7FFFu + ((u >> 16) & 1u);   // round-to-nearest-even
  return (u16)(u >> 16);
}

__device__ __forceinline__ v8f v8zero() {
  v8f z;
#pragma unroll
  for (int i = 0; i < 8; ++i) z[i] = 0.f;
  return z;
}

__device__ __forceinline__ v8f wmma_bf16(const Frag& a, const Frag& b, v8f c) {
  return __builtin_amdgcn_wmma_f32_16x16x32_bf16(false, a.v, false, b.v,
                                                 (short)0, c, false, false);
}

// ---- CDNA5 async global->LDS copy (ASYNCcnt path), manual fallback ----------
typedef __attribute__((__vector_size__(16))) int vec4i;

__device__ __forceinline__ void g2l_async_b128(const void* g, void* l) {
#if HAVE_ASYNC_LDS
  __builtin_amdgcn_global_load_async_to_lds_b128(
      (vec4i*)(uintptr_t)g, (vec4i*)l, 0, 0);
#else
  *(uint4*)l = *(const uint4*)g;
#endif
}

__device__ __forceinline__ void wait_async0() {
#if HAVE_ASYNC_LDS
#if __has_builtin(__builtin_amdgcn_s_wait_asynccnt)
  __builtin_amdgcn_s_wait_asynccnt(0);
#else
  asm volatile("s_wait_asynccnt 0x0" ::: "memory");
#endif
#endif
}

// ---- CDNA5 Tensor Data Mover: 2D bf16 tile load (TENSORcnt path) ------------
#if HAVE_TDM
typedef __attribute__((ext_vector_type(4))) u32 tdm_v4u;
typedef __attribute__((ext_vector_type(8))) int tdm_v8i;
typedef __attribute__((ext_vector_type(4))) int tdm_v4i;

__device__ __forceinline__ void tdm_load_bf16_2d(u32 lds_off, const u16* gsrc,
                                                 u32 cols, u32 rows, u32 stride) {
  u64 ga = (u64)(uintptr_t)gsrc;
  tdm_v4u g0;
  g0[0] = 1u;                                            // count=1, user D#
  g0[1] = lds_off;                                       // lds_addr
  g0[2] = (u32)ga;                                       // global_addr[31:0]
  g0[3] = (u32)((ga >> 32) & 0x01FFFFFFu) | (2u << 30);  // addr[56:32] | type=2
  const u32 td0 = 1u << 20, td1 = 1u << 20;              // generous OOB bounds
  tdm_v8i g1;
  g1[0] = (int)(1u << 16);                               // data_size=1 (2 bytes)
  g1[1] = (int)((td0 & 0xFFFFu) << 16);                  // tensor_dim0[15:0]
  g1[2] = (int)((td0 >> 16) | ((td1 & 0xFFFFu) << 16));  // td0[31:16] | td1[15:0]
  g1[3] = (int)((td1 >> 16) | (cols << 16));             // td1[31:16] | tile_dim0
  g1[4] = (int)(rows & 0xFFFFu);                         // tile_dim1, tile_dim2=0
  g1[5] = (int)stride;                                   // tensor_dim0_stride
  g1[6] = 0;
  g1[7] = 0;
  tdm_v4i gz4; gz4[0] = 0; gz4[1] = 0; gz4[2] = 0; gz4[3] = 0;
  tdm_v8i gz8;
#pragma unroll
  for (int i = 0; i < 8; ++i) gz8[i] = 0;
  __builtin_amdgcn_tensor_load_to_lds(g0, g1, gz4, gz4, gz8, 0);
}
#endif

__device__ __forceinline__ void wait_tensor0() {
#if __has_builtin(__builtin_amdgcn_s_wait_tensorcnt)
  __builtin_amdgcn_s_wait_tensorcnt(0);
#else
  asm volatile("s_wait_tensorcnt 0x0" ::: "memory");
#endif
}

// ---------------------------------------------------------------- f32 -> bf16
__global__ __launch_bounds__(256)
void cvt_bf16(const float* __restrict__ in, u16* __restrict__ out, int n) {
  int i = blockIdx.x * blockDim.x + threadIdx.x;
  int stride = gridDim.x * blockDim.x;
  for (; i < n; i += stride) out[i] = f2bf(in[i]);
}

// --------------------------------------------------- fused QKV GEMM (bf16 WMMA)
__global__ __launch_bounds__(256)
void qkv_gemm(const u16* __restrict__ Wq2, const u16* __restrict__ Wi2,
              const u16* __restrict__ Zb,  const u16* __restrict__ Ub,
              const float* __restrict__ rwb, u16* __restrict__ wheads) {
  __shared__ __align__(16) u16 As[128 * 32];   // [m][k], k contiguous
  __shared__ __align__(16) u16 Bs[128 * 32];   // [n][k], k contiguous (transposed)

  const int tid   = threadIdx.x;
  const int bz    = blockIdx.z;          // b*2 + chunk
  const int b     = bz >> 1, chunk = bz & 1;
  const int m0    = blockIdx.x * 128;
  const int n0    = blockIdx.y * 128;    // column within chunk

  const u16* Asrc[2] = { Wq2 + (size_t)chunk * QKV_M * D_MODEL,
                         Wi2 + (size_t)chunk * QKV_M * D_MODEL };
  const u16* Xsrc[2] = { Zb + (size_t)b * D_MODEL * L_TOT + chunk * 1024,
                         Ub + (size_t)b * D_MODEL * L_TOT + chunk * 1024 };

  const int w = tid >> 5, l = tid & 31;
  const int wm = (w & 3) * 32, wn = (w >> 2) * 64;
  const int lhalf = l >> 4, ln = l & 15;

  v8f acc[2][4];
#pragma unroll
  for (int mt = 0; mt < 2; ++mt)
#pragma unroll
    for (int nt = 0; nt < 4; ++nt) acc[mt][nt] = v8zero();

  for (int ph = 0; ph < 2; ++ph) {
    const u16* A = Asrc[ph];
    const u16* X = Xsrc[ph];
    for (int k0 = 0; k0 < D_MODEL; k0 += 32) {
      __syncthreads();
      { // stage A via async DMA to LDS (straight copy, 16 bf16 per thread)
        int m = tid >> 1, hh = tid & 1;
        const u16* g = A + (size_t)(m0 + m) * D_MODEL + k0 + hh * 16;
        u16* s = As + m * 32 + hh * 16;
        g2l_async_b128(g, s);
        g2l_async_b128(g + 8, s + 8);
        if (k0 + 32 < D_MODEL)
          __builtin_prefetch(g + 32, 0, 3);   // global_prefetch_b8
      }
      { // stage B transposed: X rows k (L-contig) -> Bs[n][k] packed pairs
        int kp = tid & 15, nb = tid >> 4;
        const u16* g = X + (size_t)(k0 + 2 * kp) * L_TOT + n0 + nb * 8;
        union { uint4 q; u16 s[8]; } r0, r1;
        r0.q = *(const uint4*)g;
        r1.q = *(const uint4*)(g + L_TOT);
        u32* b32 = (u32*)Bs;
#pragma unroll
        for (int j = 0; j < 8; ++j) {
          int n = nb * 8 + j;
          b32[(n * 32 + 2 * kp) >> 1] = (u32)r0.s[j] | ((u32)r1.s[j] << 16);
        }
        if (k0 + 32 < D_MODEL)
          __builtin_prefetch(g + 32 * L_TOT, 0, 3);
      }
      wait_async0();
      __syncthreads();

      Frag a[2];
#pragma unroll
      for (int mt = 0; mt < 2; ++mt) {
        const u16* s = As + (wm + mt * 16 + ln) * 32 + lhalf * 8;
        a[mt].q[0] = *(const uint4*)s;
        a[mt].q[1] = *(const uint4*)(s + 16);
      }
#pragma unroll
      for (int nt = 0; nt < 4; ++nt) {
        Frag bf;
        const u16* s = Bs + (wn + nt * 16 + ln) * 32 + lhalf * 8;
        bf.q[0] = *(const uint4*)s;
        bf.q[1] = *(const uint4*)(s + 16);
#pragma unroll
        for (int mt = 0; mt < 2; ++mt)
          acc[mt][nt] = wmma_bf16(a[mt], bf, acc[mt][nt]);
      }
    }
  }

  u16* outb = wheads + (size_t)b * QKV_M * L_TOT;
#pragma unroll
  for (int mt = 0; mt < 2; ++mt)
#pragma unroll
    for (int nt = 0; nt < 4; ++nt)
#pragma unroll
      for (int r = 0; r < 8; ++r) {
        int row = m0 + wm + mt * 16 + r + lhalf * 8;
        int col = chunk * 1024 + n0 + wn + nt * 16 + ln;
        float v = acc[mt][nt][r];
        if (row < D_MODEL) v = (v + rwb[row]) * ATTN_SCALE;  // q: bias + scale
        outb[(size_t)row * L_TOT + col] = f2bf(v);
      }
}

// ------------------------------------------------------------- flash attention
__global__ __launch_bounds__(256)
void attention(const u16* __restrict__ wheads, u16* __restrict__ attnv) {
  __shared__ __align__(16) u16 Qt[128 * 64];   // [i][d]
  __shared__ __align__(16) u16 Kt[64 * 64];    // [j][d]
  __shared__ __align__(16) u16 Vt[64 * 64];    // [d][j]
  __shared__ __align__(16) u16 Ps[8 * 16 * 64];// per-wave P tile [i][j]

  const int tid = threadIdx.x;
  const int qb = blockIdx.x, h = blockIdx.y, b = blockIdx.z;
  const int i0 = qb * 128;
  const size_t boff = (size_t)b * QKV_M * L_TOT;
  const u16* qsrc = wheads + boff + (size_t)(h * DH) * L_TOT;
  const u16* ksrc = wheads + boff + (size_t)(D_MODEL + h * DH) * L_TOT;
  const u16* vsrc = wheads + boff + (size_t)(2 * D_MODEL + h * DH) * L_TOT;

  const int w = tid >> 5, l = tid & 31;
  const int lhalf = l >> 4, ln = l & 15;

  { // stage Q transposed once (rq already biased+scaled by qkv_gemm)
    int dp = tid & 31, ib = tid >> 5;
    const u16* g = qsrc + (size_t)(2 * dp) * L_TOT + i0 + ib * 16;
    union { uint4 q[2]; u16 s[16]; } r0, r1;
    r0.q[0] = *(const uint4*)g;             r0.q[1] = *(const uint4*)(g + 8);
    r1.q[0] = *(const uint4*)(g + L_TOT);   r1.q[1] = *(const uint4*)(g + L_TOT + 8);
    u32* q32 = (u32*)Qt;
#pragma unroll
    for (int j = 0; j < 16; ++j) {
      int i = ib * 16 + j;
      q32[(i * 64 + 2 * dp) >> 1] = (u32)r0.s[j] | ((u32)r1.s[j] << 16);
    }
  }

  v8f o[4];
#pragma unroll
  for (int dt = 0; dt < 4; ++dt) o[dt] = v8zero();
  float rmax[8], rsum[8];
#pragma unroll
  for (int r = 0; r < 8; ++r) { rmax[r] = -3.0e38f; rsum[r] = 0.f; }
  u16* Pw = Ps + w * (16 * 64);

  for (int j0 = 0; j0 < L_TOT; j0 += 64) {
    __syncthreads();
    { // stage K transposed
      int dp = tid & 31, jb = tid >> 5;
      const u16* g = ksrc + (size_t)(2 * dp) * L_TOT + j0 + jb * 8;
      union { uint4 q; u16 s[8]; } r0, r1;
      r0.q = *(const uint4*)g;
      r1.q = *(const uint4*)(g + L_TOT);
      u32* k32 = (u32*)Kt;
#pragma unroll
      for (int j = 0; j < 8; ++j) {
        int jj = jb * 8 + j;
        k32[(jj * 64 + 2 * dp) >> 1] = (u32)r0.s[j] | ((u32)r1.s[j] << 16);
      }
    }
#if HAVE_TDM
    // stage V ([d][j], straight 2D tile) with the Tensor Data Mover
    if (w == 0) {
      if (l == 0) {
        tdm_load_bf16_2d((u32)(uintptr_t)(void*)Vt, vsrc + j0,
                         /*cols=*/64, /*rows=*/64, /*stride=*/L_TOT);
      }
      wait_tensor0();
    }
#else
    { // stage V straight ([d][j], j contiguous)
      int d = tid >> 2, q4 = tid & 3;
      const u16* g = vsrc + (size_t)d * L_TOT + j0 + q4 * 16;
      uint4 v0 = *(const uint4*)g, v1 = *(const uint4*)(g + 8);
      u16* s = Vt + d * 64 + q4 * 16;
      *(uint4*)s = v0; *(uint4*)(s + 8) = v1;
    }
#endif
    __syncthreads();

    // S = rq^T k : 16 queries x 64 keys per wave
    Frag aq[2];
#pragma unroll
    for (int ks = 0; ks < 2; ++ks) {
      const u16* s = Qt + (w * 16 + ln) * 64 + lhalf * 8 + ks * 32;
      aq[ks].q[0] = *(const uint4*)s;
      aq[ks].q[1] = *(const uint4*)(s + 16);
    }
    v8f sacc[4];
#pragma unroll
    for (int nt = 0; nt < 4; ++nt) sacc[nt] = v8zero();
#pragma unroll
    for (int nt = 0; nt < 4; ++nt)
#pragma unroll
      for (int ks = 0; ks < 2; ++ks) {
        Frag bk;
        const u16* s = Kt + (nt * 16 + ln) * 64 + lhalf * 8 + ks * 32;
        bk.q[0] = *(const uint4*)s; bk.q[1] = *(const uint4*)(s + 16);
        sacc[nt] = wmma_bf16(aq[ks], bk, sacc[nt]);
      }

    // online softmax (row = VGPR r; reduce across 16-lane half with shfl_xor)
#pragma unroll
    for (int r = 0; r < 8; ++r) {
      float m4 = fmaxf(fmaxf(sacc[0][r], sacc[1][r]),
                       fmaxf(sacc[2][r], sacc[3][r]));
#pragma unroll
      for (int off = 1; off < 16; off <<= 1)
        m4 = fmaxf(m4, __shfl_xor(m4, off, 32));
      float nm = fmaxf(rmax[r], m4);
      float alpha = __expf(rmax[r] - nm);
      rmax[r] = nm;
      float ps = 0.f;
#pragma unroll
      for (int nt = 0; nt < 4; ++nt) {
        float p = __expf(sacc[nt][r] - nm);
        sacc[nt][r] = p; ps += p;
      }
#pragma unroll
      for (int off = 1; off < 16; off <<= 1)
        ps += __shfl_xor(ps, off, 32);
      rsum[r] = rsum[r] * alpha + ps;
#pragma unroll
      for (int dt = 0; dt < 4; ++dt) o[dt][r] *= alpha;
    }

    // P -> per-wave LDS (re-layout C-layout -> A-fragment layout)
#pragma unroll
    for (int nt = 0; nt < 4; ++nt)
#pragma unroll
      for (int r = 0; r < 8; ++r)
        Pw[(r + lhalf * 8) * 64 + nt * 16 + ln] = f2bf(sacc[nt][r]);

    // O += P @ V^T
    Frag ap[2];
#pragma unroll
    for (int ks = 0; ks < 2; ++ks) {
      const u16* s = Pw + ln * 64 + lhalf * 8 + ks * 32;
      ap[ks].q[0] = *(const uint4*)s;
      ap[ks].q[1] = *(const uint4*)(s + 16);
    }
#pragma unroll
    for (int dt = 0; dt < 4; ++dt)
#pragma unroll
      for (int ks = 0; ks < 2; ++ks) {
        Frag bv;
        const u16* s = Vt + (dt * 16 + ln) * 64 + lhalf * 8 + ks * 32;
        bv.q[0] = *(const uint4*)s; bv.q[1] = *(const uint4*)(s + 16);
        o[dt] = wmma_bf16(ap[ks], bv, o[dt]);
      }
  }

  // finalize: divide by row sums, store attn_vec (B, 1024, 2048) bf16
#pragma unroll
  for (int r = 0; r < 8; ++r) {
    float inv = 1.f / rsum[r];
    int i = i0 + w * 16 + r + lhalf * 8;
#pragma unroll
    for (int dt = 0; dt < 4; ++dt) {
      int d = dt * 16 + ln;
      attnv[((size_t)b * D_MODEL + h * DH + d) * L_TOT + i] = f2bf(o[dt][r] * inv);
    }
  }
}

// ------------------------------------------- out projection + bias + residual
__global__ __launch_bounds__(256)
void out_proj(const u16* __restrict__ Wob, const u16* __restrict__ attnv,
              const float* __restrict__ bo, const float* __restrict__ z1ss,
              float* __restrict__ xbuf) {
  __shared__ __align__(16) u16 As[128 * 32];
  __shared__ __align__(16) u16 Bs[128 * 32];

  const int tid = threadIdx.x;
  const int b = blockIdx.z;
  const int m0 = blockIdx.x * 128;
  const int n0 = blockIdx.y * 128;
  const u16* X = attnv + (size_t)b * D_MODEL * L_TOT;

  const int w = tid >> 5, l = tid & 31;
  const int wm = (w & 3) * 32, wn = (w >> 2) * 64;
  const int lhalf = l >> 4, ln = l & 15;

  v8f acc[2][4];
#pragma unroll
  for (int mt = 0; mt < 2; ++mt)
#pragma unroll
    for (int nt = 0; nt < 4; ++nt) acc[mt][nt] = v8zero();

  for (int k0 = 0; k0 < D_MODEL; k0 += 32) {
    __syncthreads();
    { // A via async DMA
      int m = tid >> 1, hh = tid & 1;
      const u16* g = Wob + (size_t)(m0 + m) * D_MODEL + k0 + hh * 16;
      u16* s = As + m * 32 + hh * 16;
      g2l_async_b128(g, s);
      g2l_async_b128(g + 8, s + 8);
      if (k0 + 32 < D_MODEL) __builtin_prefetch(g + 32, 0, 3);
    }
    {
      int kp = tid & 15, nb = tid >> 4;
      const u16* g = X + (size_t)(k0 + 2 * kp) * L_TOT + n0 + nb * 8;
      union { uint4 q; u16 s[8]; } r0, r1;
      r0.q = *(const uint4*)g;
      r1.q = *(const uint4*)(g + L_TOT);
      u32* b32 = (u32*)Bs;
#pragma unroll
      for (int j = 0; j < 8; ++j) {
        int n = nb * 8 + j;
        b32[(n * 32 + 2 * kp) >> 1] = (u32)r0.s[j] | ((u32)r1.s[j] << 16);
      }
      if (k0 + 32 < D_MODEL) __builtin_prefetch(g + 32 * L_TOT, 0, 3);
    }
    wait_async0();
    __syncthreads();

    Frag a[2];
#pragma unroll
    for (int mt = 0; mt < 2; ++mt) {
      const u16* s = As + (wm + mt * 16 + ln) * 32 + lhalf * 8;
      a[mt].q[0] = *(const uint4*)s;
      a[mt].q[1] = *(const uint4*)(s + 16);
    }
#pragma unroll
    for (int nt = 0; nt < 4; ++nt) {
      Frag bf;
      const u16* s = Bs + (wn + nt * 16 + ln) * 32 + lhalf * 8;
      bf.q[0] = *(const uint4*)s;
      bf.q[1] = *(const uint4*)(s + 16);
#pragma unroll
      for (int mt = 0; mt < 2; ++mt)
        acc[mt][nt] = wmma_bf16(a[mt], bf, acc[mt][nt]);
    }
  }

  float* xb = xbuf + (size_t)b * D_MODEL * L_TOT;
  const float* zb = z1ss + (size_t)b * D_MODEL * L_TOT;
#pragma unroll
  for (int mt = 0; mt < 2; ++mt)
#pragma unroll
    for (int nt = 0; nt < 4; ++nt)
#pragma unroll
      for (int r = 0; r < 8; ++r) {
        int row = m0 + wm + mt * 16 + r + lhalf * 8;
        int col = n0 + wn + nt * 16 + ln;
        xb[(size_t)row * L_TOT + col] =
            acc[mt][nt][r] + bo[row] + zb[(size_t)row * L_TOT + col];
      }
}

// --------------------------------------------------- per-column LayerNorm
__global__ __launch_bounds__(256)
void layernorm_col(const float* __restrict__ x, float* __restrict__ out) {
  __shared__ float red[256];
  __shared__ float red2[256];
  int col = blockIdx.x;
  int b = col >> 11, lc = col & 2047;
  const float* xc = x + (size_t)b * D_MODEL * L_TOT + lc;
  int tid = threadIdx.x;
  float v[4]; float s = 0.f, s2 = 0.f;
#pragma unroll
  for (int i = 0; i < 4; ++i) {
    v[i] = xc[(size_t)(tid + i * 256) * L_TOT];
    s += v[i]; s2 += v[i] * v[i];
  }
  red[tid] = s; red2[tid] = s2; __syncthreads();
  for (int st = 128; st > 0; st >>= 1) {
    if (tid < st) { red[tid] += red[tid + st]; red2[tid] += red2[tid + st]; }
    __syncthreads();
  }
  float mean = red[0] * (1.f / 1024.f);
  float var  = red2[0] * (1.f / 1024.f) - mean * mean;
  float rs = rsqrtf(var + LN_EPS);
  float* oc = out + (size_t)b * D_MODEL * L_TOT + lc;
#pragma unroll
  for (int i = 0; i < 4; ++i)
    oc[(size_t)(tid + i * 256) * L_TOT] = (v[i] - mean) * rs;
}

// -----------------------------------------------------------------------------
extern "C" void kernel_launch(void* const* d_in, const int* in_sizes, int n_in,
                              void* d_out, int out_size, void* d_ws, size_t ws_size,
                              hipStream_t stream) {
  (void)in_sizes; (void)n_in; (void)out_size; (void)ws_size;
  const float* z1ss  = (const float*)d_in[0];
  const float* u1ss  = (const float*)d_in[1];
  const float* Wqkv0 = (const float*)d_in[2];
  const float* Wqkv1 = (const float*)d_in[3];
  const float* Winj0 = (const float*)d_in[4];
  const float* Winj1 = (const float*)d_in[5];
  const float* Wo    = (const float*)d_in[6];
  const float* bo    = (const float*)d_in[7];
  const float* rwb   = (const float*)d_in[8];

  char* ws = (char*)d_ws;
  size_t off = 0;
  auto alloc = [&](size_t bytes) -> char* {
    char* p = ws + off;
    off += (bytes + 255) & ~(size_t)255;
    return p;
  };
  const size_t NACT = (size_t)2 * D_MODEL * L_TOT;      // 4.19M
  const size_t NW   = (size_t)QKV_M * D_MODEL;          // 3.15M per matrix
  u16* zb     = (u16*)alloc(NACT * 2);
  u16* ub     = (u16*)alloc(NACT * 2);
  u16* wq     = (u16*)alloc(2 * NW * 2);                // Wqkv0 | Wqkv1
  u16* wi     = (u16*)alloc(2 * NW * 2);                // Winj0 | Winj1
  u16* wob    = (u16*)alloc((size_t)D_MODEL * D_MODEL * 2);
  u16* wheads = (u16*)alloc((size_t)2 * QKV_M * L_TOT * 2);
  u16* attnv  = (u16*)alloc(NACT * 2);
  float* xbuf = (float*)alloc(NACT * 4);

  cvt_bf16<<<1024, 256, 0, stream>>>(z1ss, zb, (int)NACT);
  cvt_bf16<<<1024, 256, 0, stream>>>(u1ss, ub, (int)NACT);
  cvt_bf16<<<1024, 256, 0, stream>>>(Wqkv0, wq,        (int)NW);
  cvt_bf16<<<1024, 256, 0, stream>>>(Wqkv1, wq + NW,   (int)NW);
  cvt_bf16<<<1024, 256, 0, stream>>>(Winj0, wi,        (int)NW);
  cvt_bf16<<<1024, 256, 0, stream>>>(Winj1, wi + NW,   (int)NW);
  cvt_bf16<<<512, 256, 0, stream>>>(Wo, wob, D_MODEL * D_MODEL);

  qkv_gemm<<<dim3(QKV_M / 128, 1024 / 128, 4), 256, 0, stream>>>(
      wq, wi, zb, ub, rwb, wheads);
  attention<<<dim3(L_TOT / 128, NH, 2), 256, 0, stream>>>(wheads, attnv);
  out_proj<<<dim3(D_MODEL / 128, L_TOT / 128, 2), 256, 0, stream>>>(
      wob, attnv, bo, z1ss, xbuf);
  layernorm_col<<<dim3(2 * L_TOT), 256, 0, stream>>>(xbuf, (float*)d_out);
}
